// MHA_8658654068955
// MI455X (gfx1250) — compile-verified, matrix operand order checked by reading, not run on hardware
//
#include <hip/hip_runtime.h>

// ---------------------------------------------------------------------------
// MHA forward for gfx1250 (MI455X): bf16 WMMA + async-to-LDS staging.
// ---------------------------------------------------------------------------

#define B_  2
#define L_  2048
#define DM_ 2048
#define H_  16
#define HD_ 128
#define BL_ (B_ * L_)          // 4096 rows for all GEMMs
#define SCALE_ 0.08838834764831845f   // 1/sqrt(128)

typedef __bf16 bf16;
typedef __attribute__((ext_vector_type(16))) __bf16 v16bf;
typedef __attribute__((ext_vector_type(8)))  float  v8f;

__device__ __forceinline__ v8f wmma_bf16(v16bf a, v16bf b, v8f c) {
  return __builtin_amdgcn_wmma_f32_16x16x32_bf16(false, a, false, b, (short)0, c,
                                                 false, false);
}

// CDNA5 async copy global->LDS, 16B per lane, tracked by ASYNCcnt.
// %0 = LDS byte address VGPR (wave-relative), %1 = 32-bit global byte offset,
// %2 = SGPR64 base (GVS addressing mode).
__device__ __forceinline__ void async_ld_b128(unsigned lds_byte, const void* sbase,
                                              unsigned glb_byte) {
  asm volatile("global_load_async_to_lds_b128 %0, %1, %2 offset:0"
               :: "v"(lds_byte), "v"(glb_byte), "s"(sbase) : "memory");
}
__device__ __forceinline__ unsigned lds_addr(const void* p) {
  return (unsigned)(uintptr_t)p;   // LDS aperture is in the high 32 bits
}

// 16x32 bf16 "A-style" fragment (also used for B^T tiles staged N-major).
// lanes 0-15: rows M=0..15, K=0..7 (VGPR0-3) / K=16..23 (VGPR4-7);
// lanes 16-31: same rows, K=8..15 / K=24..31.
__device__ __forceinline__ v16bf load_frag16x32(const bf16* p, int stride) {
  const int lane = threadIdx.x & 31;
  const int r  = lane & 15;
  const int hi = lane >> 4;
  union { v16bf v; uint4 q[2]; } f;
  const bf16* row = p + (size_t)r * stride + hi * 8;
  f.q[0] = *reinterpret_cast<const uint4*>(row);
  f.q[1] = *reinterpret_cast<const uint4*>(row + 16);
  return f.v;
}

__device__ __forceinline__ void zero_v8f(v8f& v) {
#pragma unroll
  for (int r = 0; r < 8; ++r) v[r] = 0.0f;
}

// ---------------------------------------------------------------------------
// f32 -> bf16 convert
// ---------------------------------------------------------------------------
__global__ __launch_bounds__(256) void cvt_bf16_kernel(const float* __restrict__ in,
                                                       bf16* __restrict__ out, int n) {
  int i = blockIdx.x * 256 + threadIdx.x;
  if (i < n) out[i] = (bf16)in[i];
}

// ---------------------------------------------------------------------------
// Y[M,N] = A[M,K] @ Bt[N,K]^T.  Block tile 64(M) x 256(N), K-step 32.
// 8 waves as 2x4; each wave 32x64 = 2x4 WMMA tiles (8 wmma / 6 frag loads).
// Double-buffered LDS filled with global_load_async_to_lds_b128.
// ---------------------------------------------------------------------------
template <bool OUT_F32>
__global__ __launch_bounds__(256)
void gemm_bt_kernel(const bf16* __restrict__ A, const bf16* __restrict__ Bt,
                    float* __restrict__ Cf, bf16* __restrict__ Cb,
                    int M, int N, int K) {
  __shared__ __align__(16) bf16 As[2][64 * 32];    //  8 KB
  __shared__ __align__(16) bf16 Bs[2][256 * 32];   // 32 KB
  const int tid  = threadIdx.x;
  const int lane = tid & 31;
  const int w    = tid >> 5;
  const int wm   = (w >> 2) * 32;     // 0 / 32
  const int wn   = (w & 3) * 64;      // 0..192
  const int m0   = blockIdx.y * 64;
  const int n0   = blockIdx.x * 256;

  const unsigned asB[2] = { lds_addr(&As[0][0]), lds_addr(&As[1][0]) };
  const unsigned bsB[2] = { lds_addr(&Bs[0][0]), lds_addr(&Bs[1][0]) };
  const int arow = tid >> 2, ac = (tid & 3) * 8;   // A: 8 bf16 per thread

  auto issue = [&](int k0, int buf) {
    async_ld_b128(asB[buf] + (unsigned)(arow * 32 + ac) * 2, A,
                  (unsigned)((m0 + arow) * K + k0 + ac) * 2);
#pragma unroll
    for (int i = 0; i < 4; ++i) {                  // B: 4 chunks per thread
      int c = tid * 4 + i, row = c >> 2, cc = (c & 3) * 8;
      async_ld_b128(bsB[buf] + (unsigned)(row * 32 + cc) * 2, Bt,
                    (unsigned)((n0 + row) * K + k0 + cc) * 2);
    }
  };

  v8f acc[2][4];
#pragma unroll
  for (int i = 0; i < 2; ++i)
#pragma unroll
    for (int j = 0; j < 4; ++j) zero_v8f(acc[i][j]);

  const int nt = K / 32;
  issue(0, 0);
  for (int t = 0; t < nt; ++t) {
    if (t + 1 < nt) {               // prefetch next tile into the other buffer
      issue((t + 1) * 32, (t + 1) & 1);
      asm volatile("s_wait_asynccnt 5" ::: "memory");   // tile t landed; 5 in flight
    } else {
      asm volatile("s_wait_asynccnt 0" ::: "memory");
    }
    __syncthreads();

    const bf16* as = &As[t & 1][0];
    const bf16* bs = &Bs[t & 1][0];
    v16bf a0 = load_frag16x32(as + (size_t)wm * 32, 32);
    v16bf a1 = load_frag16x32(as + (size_t)(wm + 16) * 32, 32);
#pragma unroll
    for (int j = 0; j < 4; ++j) {
      v16bf bj = load_frag16x32(bs + (size_t)(wn + j * 16) * 32, 32);
      acc[0][j] = wmma_bf16(a0, bj, acc[0][j]);
      acc[1][j] = wmma_bf16(a1, bj, acc[1][j]);
    }
    __syncthreads();                // readers done before buffer is refilled
  }

  // C/D layout: lanes 0-15 -> N=lane, M=r; lanes 16-31 -> N=lane-16, M=r+8
  const int cn = lane & 15, chi = lane >> 4;
#pragma unroll
  for (int i = 0; i < 2; ++i)
#pragma unroll
    for (int j = 0; j < 4; ++j)
#pragma unroll
      for (int r = 0; r < 8; ++r) {
        int m = m0 + wm + i * 16 + chi * 8 + r;
        int n = n0 + wn + j * 16 + cn;
        float v = acc[i][j][r];
        if (OUT_F32) Cf[(size_t)m * N + n] = v;
        else         Cb[(size_t)m * N + n] = (bf16)v;
      }
}

// ---------------------------------------------------------------------------
// RoPE (faithful to reference bug: cos/sin indexed by HEAD, not position),
// fold 1/sqrt(HD) into Q, reshape to [B,H,L,HD] and V -> V^T [B,H,HD,L].
// ---------------------------------------------------------------------------
__global__ __launch_bounds__(256)
void rope_split_kernel(const bf16* __restrict__ Yq, const bf16* __restrict__ Yk,
                       const bf16* __restrict__ Yv,
                       const float* __restrict__ cosc, const float* __restrict__ sinc,
                       bf16* __restrict__ Qh, bf16* __restrict__ Kh,
                       bf16* __restrict__ Vt) {
  size_t i = (size_t)blockIdx.x * 256 + threadIdx.x;   // over B*L*DM
  int f   = (int)(i & (DM_ - 1));
  int tok = (int)(i >> 11);                            // DM_ = 2048 = 1<<11
  int h = f >> 7, d = f & 127;
  int b = tok >> 11, l = tok & (L_ - 1);               // L_ = 2048

  float c = cosc[h * HD_ + d];
  float s = sinc[h * HD_ + d];
  float qv = (float)Yq[i];
  float kv = (float)Yk[i];
  int d2 = (d < 64) ? d + 64 : d - 64;
  size_t pair = (size_t)tok * DM_ + h * HD_ + d2;
  float qr = (float)Yq[pair];
  float kr = (float)Yk[pair];
  if (d < 64) { qr = -qr; kr = -kr; }
  float qo = (qv * c + qr * s) * SCALE_;
  float ko = kv * c + kr * s;

  size_t hl = (size_t)(b * H_ + h) * L_ + l;
  Qh[hl * HD_ + d] = (bf16)qo;
  Kh[hl * HD_ + d] = (bf16)ko;
  Vt[((size_t)(b * H_ + h) * HD_ + d) * L_ + l] = Yv[i];
}

// ---------------------------------------------------------------------------
// Flash-style causal attention. Grid: (B*H, L/128). Block = 8 waves; wave w
// owns 16 query rows. Online softmax in f32; S and PV via bf16 WMMA.
// K / V^T tiles staged with async-to-LDS.
// ---------------------------------------------------------------------------
__global__ __launch_bounds__(256)
void attn_kernel(const bf16* __restrict__ Qh, const bf16* __restrict__ Kh,
                 const bf16* __restrict__ Vt, const int* __restrict__ maskp,
                 bf16* __restrict__ ctx) {
  __shared__ __align__(16) bf16 Ks[32 * 128];   // keys row-major (B^T for S)
  __shared__ __align__(16) bf16 Vs[128 * 32];   // V^T rows      (B^T for PV)
  __shared__ __align__(16) bf16 Ps[8][16 * 32]; // per-wave P re-layout buffer

  const int tid = threadIdx.x, lane = tid & 31, w = tid >> 5;
  const int bh = blockIdx.x;                 // 0..B*H-1
  const int b = bh >> 4, h = bh & 15;
  const int q0 = blockIdx.y * 128;
  const int qw = q0 + w * 16;                // wave's first query row
  const int cn = lane & 15, hi = lane >> 4;

  const bf16* Qg = Qh + (size_t)bh * L_ * HD_;
  const bf16* Kg = Kh + (size_t)bh * L_ * HD_;
  const bf16* Vg = Vt + (size_t)bh * HD_ * L_;
  const unsigned ksB = lds_addr(&Ks[0]);
  const unsigned vsB = lds_addr(&Vs[0]);

  v16bf qf[4];
#pragma unroll
  for (int j = 0; j < 4; ++j)
    qf[j] = load_frag16x32(Qg + (size_t)qw * HD_ + j * 32, HD_);

  float mrow[8], lrow[8];
  v8f o[8];
#pragma unroll
  for (int r = 0; r < 8; ++r) { mrow[r] = -3.0e38f; lrow[r] = 0.0f; }
#pragma unroll
  for (int t = 0; t < 8; ++t) zero_v8f(o[t]);

  const int nk = (q0 + 128) >> 5;            // 32-key tiles needed (causal)
  for (int kt = 0; kt < nk; ++kt) {
    const int k0 = kt << 5;
    __syncthreads();                         // previous readers done
#pragma unroll
    for (int i2 = 0; i2 < 2; ++i2) {         // async-stage K tile + V^T tile
      int c = tid * 2 + i2;                  // 0..511 chunks of 8 bf16
      int kr = c >> 4, kc = (c & 15) * 8;
      async_ld_b128(ksB + (unsigned)(kr * 128 + kc) * 2, Kg,
                    (unsigned)((k0 + kr) * HD_ + kc) * 2);
      int vr = c >> 2, vc = (c & 3) * 8;
      async_ld_b128(vsB + (unsigned)(vr * 32 + vc) * 2, Vg,
                    (unsigned)(vr * L_ + k0 + vc) * 2);
    }
    asm volatile("s_wait_asynccnt 0" ::: "memory");
    __syncthreads();

    if (k0 <= qw + 15) {                     // wave has unmasked work here
      v8f s0, s1; zero_v8f(s0); zero_v8f(s1);
#pragma unroll
      for (int j = 0; j < 4; ++j) {          // S = Q @ K^T (HD=128 = 4 x K32)
        v16bf bk0 = load_frag16x32(&Ks[0 * 128 + j * 32], 128);
        v16bf bk1 = load_frag16x32(&Ks[16 * 128 + j * 32], 128);
        s0 = wmma_bf16(qf[j], bk0, s0);
        s1 = wmma_bf16(qf[j], bk1, s1);
      }
      const int key0 = k0 + cn, key1 = k0 + 16 + cn;
      const int km0 = maskp[b * L_ + key0];
      const int km1 = maskp[b * L_ + key1];
#pragma unroll
      for (int r = 0; r < 8; ++r) {
        int qrow = qw + hi * 8 + r;
        if (key0 > qrow || km0 == 0) s0[r] = -1.0e30f;
        if (key1 > qrow || km1 == 0) s1[r] = -1.0e30f;
      }
#pragma unroll
      for (int r = 0; r < 8; ++r) {          // online softmax per row
        float t = fmaxf(s0[r], s1[r]);
#pragma unroll
        for (int off = 1; off < 16; off <<= 1) t = fmaxf(t, __shfl_xor(t, off, 32));
        float mnew  = fmaxf(mrow[r], t);
        float alpha = __expf(mrow[r] - mnew);
        float p0 = __expf(s0[r] - mnew);
        float p1 = __expf(s1[r] - mnew);
        float ps = p0 + p1;
#pragma unroll
        for (int off = 1; off < 16; off <<= 1) ps += __shfl_xor(ps, off, 32);
        lrow[r] = lrow[r] * alpha + ps;
        mrow[r] = mnew;
#pragma unroll
        for (int t2 = 0; t2 < 8; ++t2) o[t2][r] *= alpha;
        // re-layout P (C layout -> A layout) through per-wave LDS
        Ps[w][(hi * 8 + r) * 32 + cn]      = (bf16)p0;
        Ps[w][(hi * 8 + r) * 32 + 16 + cn] = (bf16)p1;
      }
      // LDS ops from one wave are in-order; compiler inserts s_wait_dscnt.
      v16bf pf = load_frag16x32(&Ps[w][0], 32);
#pragma unroll
      for (int t2 = 0; t2 < 8; ++t2) {       // O += P @ V  (128 d-cols)
        v16bf bv = load_frag16x32(&Vs[(size_t)t2 * 16 * 32], 32);
        o[t2] = wmma_bf16(pf, bv, o[t2]);
      }
    }
  }

  // normalize + write context [B, L, H*HD] (bf16, feeds final projection)
#pragma unroll
  for (int r = 0; r < 8; ++r) {
    float inv = 1.0f / lrow[r];
    int qrow = qw + hi * 8 + r;
    size_t base = ((size_t)(b * L_ + qrow)) * DM_ + h * HD_;
#pragma unroll
    for (int t2 = 0; t2 < 8; ++t2)
      ctx[base + t2 * 16 + cn] = (bf16)(o[t2][r] * inv);
  }
}

// ---------------------------------------------------------------------------
extern "C" void kernel_launch(void* const* d_in, const int* in_sizes, int n_in,
                              void* d_out, int out_size, void* d_ws, size_t ws_size,
                              hipStream_t stream) {
  const float* hidden = (const float*)d_in[0];
  const int*   maskp  = (const int*)d_in[1];
  const float* Wq = (const float*)d_in[2];
  const float* Wk = (const float*)d_in[3];
  const float* Wv = (const float*)d_in[4];
  const float* Wo = (const float*)d_in[5];
  const float* cosc = (const float*)d_in[6];
  const float* sinc = (const float*)d_in[7];
  float* out = (float*)d_out;

  // ---- workspace layout (bf16 intermediates) ----
  char* ws = (char*)d_ws;
  size_t off = 0;
  const size_t XB = (size_t)BL_ * DM_ * sizeof(bf16);   // 16 MB
  const size_t WB = (size_t)DM_ * DM_ * sizeof(bf16);   //  8 MB
  bf16* Xbf = (bf16*)(ws + off); off += XB;             // reused later as ctx
  bf16* Wqb = (bf16*)(ws + off); off += WB;
  bf16* Wkb = (bf16*)(ws + off); off += WB;
  bf16* Wvb = (bf16*)(ws + off); off += WB;
  bf16* Wob = (bf16*)(ws + off); off += WB;
  bf16* Yq  = (bf16*)(ws + off); off += XB;
  bf16* Yk  = (bf16*)(ws + off); off += XB;
  bf16* Yv  = (bf16*)(ws + off); off += XB;
  bf16* Qhb = (bf16*)(ws + off); off += XB;
  bf16* Khb = (bf16*)(ws + off); off += XB;
  bf16* Vtb = (bf16*)(ws + off); off += XB;
  bf16* ctx = Xbf;   // X no longer needed once the 3 projections are done

  const int nX = BL_ * DM_;     // 8388608
  const int nW = DM_ * DM_;     // 4194304

  // 1) converts
  cvt_bf16_kernel<<<nX / 256, 256, 0, stream>>>(hidden, Xbf, nX);
  cvt_bf16_kernel<<<nW / 256, 256, 0, stream>>>(Wq, Wqb, nW);
  cvt_bf16_kernel<<<nW / 256, 256, 0, stream>>>(Wk, Wkb, nW);
  cvt_bf16_kernel<<<nW / 256, 256, 0, stream>>>(Wv, Wvb, nW);
  cvt_bf16_kernel<<<nW / 256, 256, 0, stream>>>(Wo, Wob, nW);

  // 2) QKV projections  (Y = X @ W.T; W row-major [N][K] is exactly B^T)
  dim3 ggrid(DM_ / 256, BL_ / 64);
  gemm_bt_kernel<false><<<ggrid, 256, 0, stream>>>(Xbf, Wqb, nullptr, Yq, BL_, DM_, DM_);
  gemm_bt_kernel<false><<<ggrid, 256, 0, stream>>>(Xbf, Wkb, nullptr, Yk, BL_, DM_, DM_);
  gemm_bt_kernel<false><<<ggrid, 256, 0, stream>>>(Xbf, Wvb, nullptr, Yv, BL_, DM_, DM_);

  // 3) RoPE (head-indexed, per reference) + head split + V transpose
  rope_split_kernel<<<nX / 256, 256, 0, stream>>>(Yq, Yk, Yv, cosc, sinc,
                                                  Qhb, Khb, Vtb);

  // 4) causal flash attention
  dim3 agrid(B_ * H_, L_ / 128);
  attn_kernel<<<agrid, 256, 0, stream>>>(Qhb, Khb, Vtb, maskp, ctx);

  // 5) output projection -> f32
  gemm_bt_kernel<true><<<ggrid, 256, 0, stream>>>(ctx, Wob, out, nullptr, BL_, DM_, DM_);
}
